// MultiLayerAttention_57552561766923
// MI455X (gfx1250) — matrix operand order repaired and checked by reading, not compiled, not run on hardware
//
#include <hip/hip_runtime.h>
#include <hip/hip_bf16.h>

// ---------------------------------------------------------------------------
// Problem constants (from reference)
// ---------------------------------------------------------------------------
#define BB        2
#define SS        2048
#define HH        2048
#define N_HEADS   16
#define N_KV      4
#define GQ        (N_HEADS / N_KV)      // 4
#define DHEAD     128
#define SOFT_SCALE 0.08838834764831845f // 128^-0.5
#define MASKED_LOGIT (-393024.0f)       // MASK_VALUE(-65504) * unscale(6)

typedef __attribute__((ext_vector_type(16))) _Float16 v16h;
typedef __attribute__((ext_vector_type(8)))  _Float16 v8h;
typedef __attribute__((ext_vector_type(8)))  float    v8f;
typedef __attribute__((ext_vector_type(4)))  unsigned u32x4;
typedef __attribute__((ext_vector_type(8)))  unsigned u32x8;

union AFrag { v16h v; v8h h[2]; };

// ---------------------------------------------------------------------------
// WMMA fragment helpers (wave32, 16x16x32 f16 -> f32)
//
// A (16x32, row major src): lane L holds row M=L&15,
//   K = 16*h .. 16*h+15 where h = L>>4.
// B (32x16) built from W stored as [N][K] row-major (i.e. B[k][n] = W[n][k]):
//   lane L holds column N=L&15, K = 16*(L>>4) .. +15  -> one contiguous v16h.
// C/D (16x16): lane L, vgpr r -> element (M = r + 8*(L>>4), N = L&15).
// ---------------------------------------------------------------------------
__device__ __forceinline__ v16h load_frag_a(const _Float16* base, long ld) {
  const int lane = threadIdx.x & 31;
  const int m    = lane & 15;
  const int h8   = (lane >> 4) << 4;
  const _Float16* r = base + (long)m * ld;
  AFrag f;
  f.h[0] = *(const v8h*)(r + h8);
  f.h[1] = *(const v8h*)(r + 8 + h8);
  return f.v;
}

__device__ __forceinline__ v16h load_frag_b(const _Float16* base, long ld) {
  const int lane = threadIdx.x & 31;
  const int n    = lane & 15;
  const int k16  = (lane >> 4) << 4;
  return *(const v16h*)(base + (long)n * ld + k16);
}

__device__ __forceinline__ v8f wmma_f16(v16h a, v16h b, v8f c) {
  return __builtin_amdgcn_wmma_f32_16x16x32_f16(false, a, false, b,
                                                (short)0, c, false, false);
}

// Reductions across a 16-lane half (xor masks 1,2,4,8 stay within the half).
__device__ __forceinline__ float half_max(float v) {
  #pragma unroll
  for (int m = 8; m >= 1; m >>= 1) v = fmaxf(v, __shfl_xor(v, m, 32));
  return v;
}
__device__ __forceinline__ float half_sum(float v) {
  #pragma unroll
  for (int m = 8; m >= 1; m >>= 1) v += __shfl_xor(v, m, 32);
  return v;
}

// ---------------------------------------------------------------------------
// Tensor Data Mover: DMA a [tile_d1 x tile_d0] f16 tile (row stride = ld
// elements in global) into LDS at lds_byte_addr, inserting 4 DWORDs of pad
// every 16 DWORDs (row 32 halfs -> LDS row stride 40 halfs, bank-friendly).
// D# layout per CDNA5 ISA ch.8 (group0 128b, group1 256b; 2D -> groups 2/3
// disabled). Issued by one wave; tracked with TENSORcnt.
// ---------------------------------------------------------------------------
__device__ __forceinline__ void tdm_load_tile_f16(const _Float16* gsrc,
                                                  unsigned lds_byte_addr,
                                                  unsigned tensor_d0,
                                                  unsigned tensor_d1,
                                                  unsigned stride_d0,
                                                  unsigned tile_d0,
                                                  unsigned tile_d1) {
  unsigned long long ga = (unsigned long long)(size_t)gsrc;
  u32x4 g0;
  g0[0] = 1u;                                        // count=1, user descriptor
  g0[1] = lds_byte_addr;                             // lds_addr [63:32]
  g0[2] = (unsigned)(ga & 0xffffffffu);              // global_addr [95:64]
  g0[3] = (unsigned)((ga >> 32) & 0x01ffffffu)       // global_addr [120:96]
        | (2u << 30);                                // type = 2 ("image")
  u32x8 g1;
  g1[0] = (1u << 16)                                 // data_size: 2 bytes
        | (1u << 20)                                 // pad_enable
        | (3u << 22)                                 // pad_interval: 16 DWORDs
        | (3u << 25);                                // pad_amount: 4 DWORDs
  g1[1] = (tensor_d0 & 0xffffu) << 16;               // tensor_dim0 [79:48] lo
  g1[2] = ((tensor_d0 >> 16) & 0xffffu)              // tensor_dim0 hi
        | ((tensor_d1 & 0xffffu) << 16);             // tensor_dim1 lo
  g1[3] = ((tensor_d1 >> 16) & 0xffffu)              // tensor_dim1 hi
        | ((tile_d0 & 0xffffu) << 16);               // tile_dim0 [127:112]
  g1[4] = (tile_d1 & 0xffffu);                       // tile_dim1; tile_dim2=0
  g1[5] = stride_d0;                                 // tensor_dim0_stride lo32
  g1[6] = 0u;                                        // stride hi / dim1_stride
  g1[7] = 0u;
  asm volatile("tensor_load_to_lds %0, %1" :: "s"(g0), "s"(g1) : "memory");
}

// ---------------------------------------------------------------------------
// f32 -> f16 convert
// ---------------------------------------------------------------------------
__global__ void cvt_f32_to_f16(const float* __restrict__ in,
                               _Float16* __restrict__ out, int n) {
  int i = blockIdx.x * blockDim.x + threadIdx.x;
  int stride = gridDim.x * blockDim.x;
  for (; i < n; i += stride) out[i] = (_Float16)in[i];
}

// V [BH=8][S][D] f32  ->  Vt [8][D][S] f16  (so PV B-fragments are contiguous)
__global__ void transpose_v_f16(const float* __restrict__ V,
                                _Float16* __restrict__ Vt) {
  long total = (long)BB * N_KV * SS * DHEAD;
  long i = (long)blockIdx.x * blockDim.x + threadIdx.x;
  long stride = (long)gridDim.x * blockDim.x;
  for (; i < total; i += stride) {
    long bh  = i / ((long)SS * DHEAD);
    long rem = i % ((long)SS * DHEAD);
    long s   = rem / DHEAD;
    long d   = rem % DHEAD;
    Vt[(bh * DHEAD + d) * SS + s] = (_Float16)V[i];
  }
}

// ---------------------------------------------------------------------------
// WMMA GEMM:  out[M,N] = A[M,K] (f16, row major) x W[N,K]^T (f16) + bias[N]
// Block: 256 threads = 8 waves (2 x 4), block tile 128x128,
// wave tile 64x32 = 4x2 wmma tiles, K step 32.
// B tile (128 W-rows x 32 K, 8KB) is DMA'd into LDS by the Tensor Data Mover
// (wave 0 issues; TENSORcnt-tracked, double buffered). A fragments stream
// from global (L2-resident).
// ---------------------------------------------------------------------------
__global__ __launch_bounds__(256)
void gemm_wmma_f16(const _Float16* __restrict__ A,
                   const _Float16* __restrict__ W,
                   const float* __restrict__ bias,
                   float* __restrict__ outF,
                   _Float16* __restrict__ outH,
                   int M, int N, int K) {
  __shared__ _Float16 Bt[2][128][40];   // padded rows: 32 data + 8 pad halfs

  const int wave = threadIdx.x >> 5;
  const int lane = threadIdx.x & 31;
  const int wm = wave >> 2;       // 0..1
  const int wn = wave & 3;        // 0..3
  const long m0 = (long)blockIdx.y * 128 + wm * 64;
  const long n0blk = (long)blockIdx.x * 128;
  const long n0 = n0blk + wn * 32;

  const _Float16* Wblk = W + n0blk * (long)K;   // 128 W-rows for this block
  const unsigned ldsb[2] = { (unsigned)(size_t)&Bt[0][0][0],
                             (unsigned)(size_t)&Bt[1][0][0] };

  v8f acc[4][2];
  #pragma unroll
  for (int i = 0; i < 4; ++i)
    #pragma unroll
    for (int j = 0; j < 2; ++j) acc[i][j] = (v8f)(0.0f);

  const int niter = K >> 5;                     // K/32 tiles
  if (wave == 0)                                // prologue: tile 0 -> buf 0
    tdm_load_tile_f16(Wblk, ldsb[0], K, 128, K, 32, 128);

  for (int it = 0; it < niter; ++it) {
    if (wave == 0) {
      if (it + 1 < niter) {
        tdm_load_tile_f16(Wblk + (long)(it + 1) * 32, ldsb[(it + 1) & 1],
                          K, 128, K, 32, 128);
        asm volatile("s_wait_tensorcnt 0x1" ::: "memory"); // tile `it` resident
      } else {
        asm volatile("s_wait_tensorcnt 0x0" ::: "memory");
      }
    }
    __syncthreads();                            // B tile visible to all waves

    const int kk = it << 5;
    const _Float16* Bl = &Bt[it & 1][0][0];

    v16h a[4], b[2];
    #pragma unroll
    for (int i = 0; i < 4; ++i)
      a[i] = load_frag_a(A + (m0 + 16 * i) * (long)K + kk, K);
    #pragma unroll
    for (int j = 0; j < 2; ++j)
      b[j] = load_frag_b(Bl + (wn * 32 + 16 * j) * 40, 40);  // ds_load_b128 x2
    #pragma unroll
    for (int i = 0; i < 4; ++i)
      #pragma unroll
      for (int j = 0; j < 2; ++j)
        acc[i][j] = wmma_f16(a[i], b[j], acc[i][j]);

    __syncthreads();                            // done reading before refill
  }

  const int n  = lane & 15;
  const int mh = (lane >> 4) << 3;
  #pragma unroll
  for (int j = 0; j < 2; ++j) {
    const long col = n0 + 16 * j + n;
    const float bv = bias[col];
    #pragma unroll
    for (int i = 0; i < 4; ++i) {
      #pragma unroll
      for (int r = 0; r < 8; ++r) {
        const long row = m0 + 16 * i + mh + r;
        const float o = acc[i][j][r] + bv;
        if (outF) outF[row * (long)N + col] = o;
        else      outH[row * (long)N + col] = (_Float16)o;
      }
    }
  }
}

// ---------------------------------------------------------------------------
// Fused causal flash attention (GQA), f16 WMMA, fp32 online softmax.
// Grid: (S/128, B*N_HEADS). Block 256 = 8 waves, each wave owns 16 q rows.
// Qh: [B,S,H] f16   Kh: [B,KV,S,D] f16   Vt: [B,KV,D,S] f16
// Ah: [B,S,H] f16 attention output (pre out-projection)
// ---------------------------------------------------------------------------
__global__ __launch_bounds__(256)
void attn_flash_wmma(const _Float16* __restrict__ Qh,
                     const _Float16* __restrict__ Kh,
                     const _Float16* __restrict__ Vt,
                     _Float16* __restrict__ Ah) {
  __shared__ _Float16 Plds[8][16][40];   // per-wave P staging, padded rows

  const int wave = threadIdx.x >> 5;
  const int lane = threadIdx.x & 31;
  const int n    = lane & 15;
  const int mh   = (lane >> 4) << 3;

  const int head = blockIdx.y;                // 0..31
  const int b    = head / N_HEADS;
  const int h    = head % N_HEADS;
  const int kv   = h / GQ;
  const int q0   = blockIdx.x * 128 + wave * 16;

  const _Float16* Qbase = Qh + ((long)(b * SS + q0)) * HH + h * DHEAD;
  const _Float16* Kbase = Kh + ((long)(b * N_KV + kv)) * SS * DHEAD;
  const _Float16* Vbase = Vt + ((long)(b * N_KV + kv)) * DHEAD * SS;

  // Q fragments covering d = 0..127 (4 x 16x32 A-fragments)
  v16h qf[4];
  #pragma unroll
  for (int c = 0; c < 4; ++c) qf[c] = load_frag_a(Qbase + 32 * c, HH);

  v8f acc[8];
  #pragma unroll
  for (int t = 0; t < 8; ++t) acc[t] = (v8f)(0.0f);
  float mi[8], li[8];
  #pragma unroll
  for (int r = 0; r < 8; ++r) { mi[r] = -1e30f; li[r] = 0.0f; }

  const int nkt = (q0 + 16 + 31) >> 5;        // 32-key tiles covering causal span
  for (int kt = 0; kt < nkt; ++kt) {
    const int s0 = kt << 5;
    const bool need_mask = (s0 + 31 > q0);

    // ---- scores: two 16x16 tiles (keys s0..s0+15 and s0+16..s0+31)
    v8f sc0 = (v8f)(0.0f), sc1 = (v8f)(0.0f);
    #pragma unroll
    for (int c = 0; c < 4; ++c) {
      v16h kb0 = load_frag_b(Kbase + (long)(s0)      * DHEAD + 32 * c, DHEAD);
      v16h kb1 = load_frag_b(Kbase + (long)(s0 + 16) * DHEAD + 32 * c, DHEAD);
      sc0 = wmma_f16(qf[c], kb0, sc0);
      sc1 = wmma_f16(qf[c], kb1, sc1);
    }

    // ---- scale, mask, online softmax (row r+mh lives in this lane-half)
    v8f p0, p1;
    #pragma unroll
    for (int r = 0; r < 8; ++r) {
      const int qg = q0 + mh + r;
      float v0 = sc0[r] * SOFT_SCALE;
      float v1 = sc1[r] * SOFT_SCALE;
      if (need_mask) {
        if (s0 + n      > qg) v0 = MASKED_LOGIT;
        if (s0 + 16 + n > qg) v1 = MASKED_LOGIT;
      }
      const float tmax = half_max(fmaxf(v0, v1));
      const float mnew = fmaxf(mi[r], tmax);
      const float corr = __expf(mi[r] - mnew);
      const float e0 = __expf(v0 - mnew);
      const float e1 = __expf(v1 - mnew);
      li[r] = li[r] * corr + half_sum(e0 + e1);
      mi[r] = mnew;
      p0[r] = e0;
      p1[r] = e1;
      #pragma unroll
      for (int t = 0; t < 8; ++t) acc[t][r] *= corr;
    }

    // ---- stage P (C-layout) into LDS, reload as A-layout 16x32 fragment
    #pragma unroll
    for (int r = 0; r < 8; ++r) {
      Plds[wave][mh + r][n]      = (_Float16)p0[r];
      Plds[wave][mh + r][16 + n] = (_Float16)p1[r];
    }
    asm volatile("s_wait_dscnt 0" ::: "memory");
    v16h pa = load_frag_a(&Plds[wave][0][0], 40);

    // ---- out += P(16x32) x V(32x128): 8 column tiles of 16
    #pragma unroll
    for (int nt = 0; nt < 8; ++nt) {
      v16h vb = load_frag_b(Vbase + (long)(nt * 16) * SS + s0, SS);
      acc[nt] = wmma_f16(pa, vb, acc[nt]);
    }
  }

  // ---- epilogue: normalize and store f16 attention output
  #pragma unroll
  for (int r = 0; r < 8; ++r) {
    const float inv = 1.0f / li[r];
    const long row = (long)(b * SS + q0 + mh + r);
    #pragma unroll
    for (int nt = 0; nt < 8; ++nt) {
      const float o = acc[nt][r] * inv;
      Ah[row * HH + h * DHEAD + nt * 16 + n] = (_Float16)o;
    }
  }
}

// ---------------------------------------------------------------------------
// Host orchestration
// ---------------------------------------------------------------------------
extern "C" void kernel_launch(void* const* d_in, const int* in_sizes, int n_in,
                              void* d_out, int out_size, void* d_ws, size_t ws_size,
                              hipStream_t stream) {
  const float* hidden = (const float*)d_in[0];   // [B,S,H]
  const float* key    = (const float*)d_in[1];   // [B,KV,S,D]
  const float* value  = (const float*)d_in[2];   // [B,KV,S,D]
  const float* w_q    = (const float*)d_in[3];   // [H,H]
  const float* b_q    = (const float*)d_in[4];   // [H]
  const float* w_proj = (const float*)d_in[5];   // [H,H]
  const float* b_proj = (const float*)d_in[6];   // [H]
  float* out = (float*)d_out;                    // [B,S,H]

  char* w = (char*)d_ws;
  const long MB = 1l << 20;
  _Float16* Xh  = (_Float16*)(w + 0 * MB);   // 16 MB  [B*S, H]
  _Float16* Wqh = (_Float16*)(w + 16 * MB);  //  8 MB  [H, H]
  _Float16* Wph = (_Float16*)(w + 24 * MB);  //  8 MB  [H, H]
  _Float16* Kh  = (_Float16*)(w + 32 * MB);  //  4 MB  [B,KV,S,D]
  _Float16* Vth = (_Float16*)(w + 36 * MB);  //  4 MB  [B,KV,D,S]
  _Float16* Qh  = (_Float16*)(w + 40 * MB);  // 16 MB  [B*S, H]
  _Float16* Ah  = (_Float16*)(w + 56 * MB);  // 16 MB  [B*S, H]

  const int nX = BB * SS * HH;            // 8388608
  const int nW = HH * HH;                 // 4194304
  const int nKV = BB * N_KV * SS * DHEAD; // 2097152

  const int CB = 256;
  cvt_f32_to_f16<<<(nX + CB - 1) / CB, CB, 0, stream>>>(hidden, Xh, nX);
  cvt_f32_to_f16<<<(nW + CB - 1) / CB, CB, 0, stream>>>(w_q, Wqh, nW);
  cvt_f32_to_f16<<<(nW + CB - 1) / CB, CB, 0, stream>>>(w_proj, Wph, nW);
  cvt_f32_to_f16<<<(nKV + CB - 1) / CB, CB, 0, stream>>>(key, Kh, nKV);
  transpose_v_f16<<<(nKV + CB - 1) / CB, CB, 0, stream>>>(value, Vth);

  // Q = X @ Wq^T + b_q  (f16 result for attention)
  dim3 gGemm(HH / 128, (BB * SS) / 128);
  gemm_wmma_f16<<<gGemm, 256, 0, stream>>>(Xh, Wqh, b_q, nullptr, Qh,
                                           BB * SS, HH, HH);

  // fused causal GQA flash attention
  dim3 gAttn(SS / 128, BB * N_HEADS);
  attn_flash_wmma<<<gAttn, 256, 0, stream>>>(Qh, Kh, Vth, Ah);

  // out = Attn @ Wp^T + b_proj  (f32 result)
  gemm_wmma_f16<<<gGemm, 256, 0, stream>>>(Ah, Wph, b_proj, out, nullptr,
                                           BB * SS, HH, HH);
}
